// VisionBlock_65180423685512
// MI455X (gfx1250) — matrix operand order repaired and checked by reading, not compiled, not run on hardware
//
#include <hip/hip_runtime.h>
#include <hip/hip_bf16.h>
#include <math.h>

// ---------------------------------------------------------------------------
// Vision block (CPE + LN + decomposed global attention w/ RoPE + LEPE + FFN)
// MI455X / gfx1250: wave32, WMMA f32_16x16x32_f16 for all matrix math.
// Round 2: parallel softmax, double-buffered GEMM LDS, bank-conflict padding,
//          transposed V staging so all WMMA fragments vectorize to ds_load_b128.
// ---------------------------------------------------------------------------

typedef __attribute__((ext_vector_type(16))) _Float16 v16h;
typedef __attribute__((ext_vector_type(8)))  float    v8f;

union FragH { v16h v; _Float16 h[16]; };
union FragF { v8f  v; float    f[8];  };
union U4H   { uint4 u; _Float16 h[8]; };

__device__ __forceinline__ v8f wmma16x16x32(v16h a, v16h b, v8f c) {
  // D = A(16x32,f16) * B(32x16,f16) + C(16x16,f32)
  return __builtin_amdgcn_wmma_f32_16x16x32_f16(
      /*neg_a=*/false, a, /*neg_b=*/false, b,
      /*c_mod=*/(short)0, c, /*reuse_a=*/false, /*reuse_b=*/false);
}

// 16-bit A 16x32, K-contiguous to match B: lane<16 K=0..15, lane>=16 K=16..31.
// half i -> K = i + 16*(lane/16)
__device__ __forceinline__ int kmapA(int i, int lane) {
  return i + ((lane >> 4) << 4);
}
// B 32x16 (K-contiguous, same order as A): lane<16 K=0..15, lane>=16 K=16..31.
__device__ __forceinline__ int kmapB(int i, int lane) {
  return i + ((lane >> 4) << 4);
}

// A fragment from row-major [m][k] storage (ld = row stride in halves).
// halves 0..7 and 8..15 are contiguous 16B runs -> 2x ds_load_b128.
__device__ __forceinline__ FragH fragA_rm(const _Float16* base, int m0, int ld, int lane) {
  FragH f; int m = m0 + (lane & 15);
#pragma unroll
  for (int i = 0; i < 16; ++i) f.h[i] = base[m * ld + kmapA(i, lane)];
  return f;
}
// B fragment where storage is [n][k]; halves 0..15 contiguous -> 2x ds_load_b128.
__device__ __forceinline__ FragH fragB_nk(const _Float16* base, int n0, int ld, int lane) {
  FragH f; int n = n0 + (lane & 15);
#pragma unroll
  for (int i = 0; i < 16; ++i) f.h[i] = base[n * ld + kmapB(i, lane)];
  return f;
}

// ---------------------------------------------------------------------------
// Small utility kernels
// ---------------------------------------------------------------------------
__global__ void k_convert(const float* __restrict__ src, _Float16* __restrict__ dst,
                          int n, float scale) {
  int i = blockIdx.x * 256 + threadIdx.x;
  if (i < n) dst[i] = (_Float16)(src[i] * scale);
}

// x + dwconv3x3(x) over [8,56,56,256]
__global__ void k_cpe(const float* __restrict__ x, const float* __restrict__ w,
                      const float* __restrict__ b, float* __restrict__ out) {
  int idx = blockIdx.x * 256 + threadIdx.x;
  int c = idx & 255; int t = idx >> 8;
  int ww = t % 56; int hh = (t / 56) % 56; int bb = t / 3136;
  float acc = b[c];
#pragma unroll
  for (int kh = 0; kh < 3; ++kh) {
    int h2 = hh + kh - 1; if (h2 < 0 || h2 >= 56) continue;
#pragma unroll
    for (int kw = 0; kw < 3; ++kw) {
      int w2 = ww + kw - 1; if (w2 < 0 || w2 >= 56) continue;
      acc += x[((bb * 56 + h2) * 56 + w2) * 256 + c] * w[(kh * 3 + kw) * 256 + c];
    }
  }
  out[idx] = x[idx] + acc;
}

// LayerNorm over C=256, one wave per token, f16 output
__global__ void k_ln(const float* __restrict__ x, const float* __restrict__ g,
                     const float* __restrict__ be, _Float16* __restrict__ out, int ntok) {
  int lane = threadIdx.x & 31; int wv = threadIdx.x >> 5;
  int tok = blockIdx.x * 8 + wv;
  if (tok >= ntok) return;
  const float* xp = x + (size_t)tok * 256;
  float v[8]; float s = 0.f;
#pragma unroll
  for (int j = 0; j < 8; ++j) { v[j] = xp[lane + 32 * j]; s += v[j]; }
#pragma unroll
  for (int o = 16; o > 0; o >>= 1) s += __shfl_xor(s, o, 32);
  float mu = s * (1.f / 256.f);
  float vs = 0.f;
#pragma unroll
  for (int j = 0; j < 8; ++j) { float d = v[j] - mu; vs += d * d; }
#pragma unroll
  for (int o = 16; o > 0; o >>= 1) vs += __shfl_xor(vs, o, 32);
  float rs = rsqrtf(vs * (1.f / 256.f) + 1e-6f);
#pragma unroll
  for (int j = 0; j < 8; ++j) {
    int c = lane + 32 * j;
    out[(size_t)tok * 256 + c] = (_Float16)((v[j] - mu) * rs * g[c] + be[c]);
  }
}

// RoPE on q,k + head-major reshape, plus v copy. One thread per (b,n,h,w,pair)
__global__ void k_rope(const float* __restrict__ q, const float* __restrict__ k,
                       const float* __restrict__ v, _Float16* __restrict__ qrh,
                       _Float16* __restrict__ krh, _Float16* __restrict__ vh) {
  int idx = blockIdx.x * 256 + threadIdx.x;       // total 8*8*56*56*16
  int j = idx & 15; int t = idx >> 4;
  int w = t % 56; t /= 56;
  int h = t % 56; t /= 56;
  int n = t % 8;  int b = t / 8;
  int src = ((b * 56 + h) * 56 + w) * 256 + n * 32 + 2 * j;
  int dst = (((b * 8 + n) * 56 + h) * 56 + w) * 32 + 2 * j;
  float ang = powf(10000.f, -(float)j * (1.f / 15.f));
  float ph = (float)(h * 56 + w) * ang;
  float sn = sinf(ph), cs = cosf(ph);
  float q1 = q[src], q2 = q[src + 1];
  qrh[dst] = (_Float16)(q1 * cs - q2 * sn);
  qrh[dst + 1] = (_Float16)(q2 * cs + q1 * sn);
  float k1 = k[src], k2 = k[src + 1];
  krh[dst] = (_Float16)(k1 * cs - k2 * sn);
  krh[dst + 1] = (_Float16)(k2 * cs + k1 * sn);
  vh[dst] = (_Float16)v[src];
  vh[dst + 1] = (_Float16)v[src + 1];
}

// aoh = f16( ao + dwconv5x5(v) )   (LEPE + add, f16 out for Wo GEMM)
__global__ void k_lepe_add(const float* __restrict__ ao, const float* __restrict__ v,
                           const float* __restrict__ w, const float* __restrict__ b,
                           _Float16* __restrict__ out) {
  int idx = blockIdx.x * 256 + threadIdx.x;
  int c = idx & 255; int t = idx >> 8;
  int ww = t % 56; int hh = (t / 56) % 56; int bb = t / 3136;
  float acc = b[c];
#pragma unroll
  for (int kh = 0; kh < 5; ++kh) {
    int h2 = hh + kh - 2; if (h2 < 0 || h2 >= 56) continue;
#pragma unroll
    for (int kw = 0; kw < 5; ++kw) {
      int w2 = ww + kw - 2; if (w2 < 0 || w2 >= 56) continue;
      acc += v[((bb * 56 + h2) * 56 + w2) * 256 + c] * w[(kh * 5 + kw) * 256 + c];
    }
  }
  out[idx] = (_Float16)(ao[idx] + acc);
}

// hdnh = f16( dwconv3x3(hdn) + dw_b + hdn )  over FFN=1024 channels
__global__ void k_dwadd(const float* __restrict__ x, const float* __restrict__ w,
                        const float* __restrict__ b, _Float16* __restrict__ out) {
  int idx = blockIdx.x * 256 + threadIdx.x;
  int c = idx & 1023; int t = idx >> 10;
  int ww = t % 56; int hh = (t / 56) % 56; int bb = t / 3136;
  float acc = b[c];
#pragma unroll
  for (int kh = 0; kh < 3; ++kh) {
    int h2 = hh + kh - 1; if (h2 < 0 || h2 >= 56) continue;
#pragma unroll
    for (int kw = 0; kw < 3; ++kw) {
      int w2 = ww + kw - 1; if (w2 < 0 || w2 >= 56) continue;
      acc += x[((bb * 56 + h2) * 56 + w2) * 1024 + c] * w[(kh * 3 + kw) * 1024 + c];
    }
  }
  out[idx] = (_Float16)(acc + x[idx]);
}

// ---------------------------------------------------------------------------
// Tiled WMMA GEMM: out[M,Nout] = A[M,K](f16) * W[Nout,K]^T(f16) + bias
// MODE 0: plain   1: + residual   2: exact GELU
// Block: 256 threads (8 waves), 64x64 tile, K double-buffered 32 at a time.
// LDS rows padded to 40 halves (80B) -> conflict-free ds_load_b128 fragments.
// ---------------------------------------------------------------------------
#define GLD 40
template <int MODE>
__global__ void k_gemm(const _Float16* __restrict__ A, const _Float16* __restrict__ Bw,
                       const float* __restrict__ bias, float bias_scale,
                       const float* __restrict__ residual, float* __restrict__ out,
                       int M, int Nout, int K) {
  __shared__ _Float16 sA[2][64 * GLD];
  __shared__ _Float16 sB[2][64 * GLD];
  int t = threadIdx.x;
  int lane = t & 31, wv = t >> 5;
  int r = t >> 2, c8 = (t & 3) * 8;
  int m_sub = wv & 3;            // 0..3
  int n_base = (wv >> 2) * 32;   // 0 or 32; covers n_base, n_base+16

  FragF acc0, acc1;
#pragma unroll
  for (int i = 0; i < 8; ++i) { acc0.f[i] = 0.f; acc1.f[i] = 0.f; }

  const _Float16* Arow = A + (size_t)(blockIdx.x * 64 + r) * K;
  const _Float16* Brow = Bw + (size_t)(blockIdx.y * 64 + r) * K;
  int sidx = r * GLD + c8;

  // stage slab 0 into buffer 0
  *(uint4*)(&sA[0][sidx]) = *(const uint4*)(Arow + c8);
  *(uint4*)(&sB[0][sidx]) = *(const uint4*)(Brow + c8);

  int p = 0;
  for (int k0 = 0; k0 < K; k0 += 32) {
    __syncthreads();
    if (k0 + 32 < K) {           // stage next slab into the other buffer
      *(uint4*)(&sA[p ^ 1][sidx]) = *(const uint4*)(Arow + k0 + 32 + c8);
      *(uint4*)(&sB[p ^ 1][sidx]) = *(const uint4*)(Brow + k0 + 32 + c8);
      if (k0 + 64 < K) {         // global_prefetch_b8 two slabs ahead
        __builtin_prefetch(Arow + k0 + 64 + c8, 0, 1);
        __builtin_prefetch(Brow + k0 + 64 + c8, 0, 1);
      }
    }
    FragH a  = fragA_rm(sA[p], m_sub * 16, GLD, lane);
    FragH b0 = fragB_nk(sB[p], n_base, GLD, lane);
    FragH b1 = fragB_nk(sB[p], n_base + 16, GLD, lane);
    acc0.v = wmma16x16x32(a.v, b0.v, acc0.v);
    acc1.v = wmma16x16x32(a.v, b1.v, acc1.v);
    p ^= 1;
  }

#pragma unroll
  for (int half = 0; half < 2; ++half) {
    FragF* acc = half ? &acc1 : &acc0;
    int nn0 = blockIdx.y * 64 + n_base + half * 16 + (lane & 15);
    float bz = bias[nn0] * bias_scale;
#pragma unroll
    for (int rr = 0; rr < 8; ++rr) {
      int gm = blockIdx.x * 64 + m_sub * 16 + rr + ((lane >> 4) << 3);
      float val = acc->f[rr] + bz;
      if (MODE == 2) val = 0.5f * val * (1.f + erff(val * 0.70710678118654752f));
      size_t o = (size_t)gm * Nout + nn0;
      if (MODE == 1) val += residual[o];
      out[o] = val;
    }
  }
}

// ---------------------------------------------------------------------------
// Shared attention core. Q/K tiles 64x32 (rows padded to 40), V staged
// TRANSPOSED as Vt[32][64] (row stride 72) so P@V B-fragments vectorize.
// S fp32 64x64, P f16 (stride 72). Softmax: one wave per 8 rows, lanes over j.
// ---------------------------------------------------------------------------
#define ALD 40
#define PLD 72

template <bool COL>
__global__ void k_attn(const _Float16* __restrict__ qrh, const _Float16* __restrict__ krh,
                       const _Float16* __restrict__ vsrc, _Float16* __restrict__ vwh,
                       float* __restrict__ ao) {
  __shared__ _Float16 sQ[64 * ALD], sK[64 * ALD];
  __shared__ _Float16 sVt[32 * PLD];               // transposed V: [d][j]
  __shared__ float sS[64 * 64];
  __shared__ _Float16 sP[64 * PLD];
  int t = threadIdx.x, lane = t & 31, wv = t >> 5;
  int r = t >> 2, c8 = (t & 3) * 8;

  size_t base; size_t rstr; int w = 0, bn = 0;
  if (COL) {
    w = blockIdx.x % 56; bn = blockIdx.x / 56;     // bn = b*8 + n
    base = (size_t)bn * (56 * 56 * 32) + (size_t)w * 32;
    rstr = 56 * 32;
  } else {
    base = (size_t)blockIdx.x * (56 * 32);
    rstr = 32;
  }

  uint4 z = make_uint4(0u, 0u, 0u, 0u);
  uint4 vq = (r < 56) ? *(const uint4*)(qrh + base + (size_t)r * rstr + c8) : z;
  uint4 vk = (r < 56) ? *(const uint4*)(krh + base + (size_t)r * rstr + c8) : z;
  uint4 vv = (r < 56) ? *(const uint4*)(vsrc + base + (size_t)r * rstr + c8) : z;
  *(uint4*)(&sQ[r * ALD + c8]) = vq;
  *(uint4*)(&sK[r * ALD + c8]) = vk;
  U4H uv; uv.u = vv;                               // scatter-transpose V
#pragma unroll
  for (int i = 0; i < 8; ++i) sVt[(c8 + i) * PLD + r] = uv.h[i];
  __syncthreads();

  // S = Q K^T  (16 tiles over 8 waves)
#pragma unroll
  for (int rep = 0; rep < 2; ++rep) {
    int id = wv + rep * 8, tm = id >> 2, tn = id & 3;
    FragH a = fragA_rm(sQ, tm * 16, ALD, lane);
    FragH b = fragB_nk(sK, tn * 16, ALD, lane);    // B[k][j] = Kr[j][k]
    FragF acc;
#pragma unroll
    for (int i = 0; i < 8; ++i) acc.f[i] = 0.f;
    acc.v = wmma16x16x32(a.v, b.v, acc.v);
#pragma unroll
    for (int rr = 0; rr < 8; ++rr) {
      int m = tm * 16 + rr + ((lane >> 4) << 3);
      int nn = tn * 16 + (lane & 15);
      sS[m * 64 + nn] = acc.f[rr];
    }
  }
  __syncthreads();

  // softmax: wave wv handles rows 8*wv .. 8*wv+7; lanes split 64 columns
#pragma unroll
  for (int rr2 = 0; rr2 < 8; ++rr2) {
    int row = wv * 8 + rr2;
    if (row < 56) {
      float v0 = (lane < 56) ? sS[row * 64 + lane] : -1e30f;
      float v1 = (lane + 32 < 56) ? sS[row * 64 + lane + 32] : -1e30f;
      float mx = fmaxf(v0, v1);
#pragma unroll
      for (int o = 16; o > 0; o >>= 1) mx = fmaxf(mx, __shfl_xor(mx, o, 32));
      float e0 = (lane < 56) ? __expf(v0 - mx) : 0.f;
      float e1 = (lane + 32 < 56) ? __expf(v1 - mx) : 0.f;
      float s = e0 + e1;
#pragma unroll
      for (int o = 16; o > 0; o >>= 1) s += __shfl_xor(s, o, 32);
      float inv = 1.f / s;
      sP[row * PLD + lane] = (_Float16)(e0 * inv);
      sP[row * PLD + lane + 32] = (_Float16)(e1 * inv);
    } else {
      sP[row * PLD + lane] = (_Float16)0.f;
      sP[row * PLD + lane + 32] = (_Float16)0.f;
    }
  }
  __syncthreads();

  // out64x32 = P(64x64) @ V(64x32); B from transposed sVt -> contiguous frags
  int tm = wv & 3, tn = wv >> 2;
  FragF acc;
#pragma unroll
  for (int i = 0; i < 8; ++i) acc.f[i] = 0.f;
#pragma unroll
  for (int kk = 0; kk < 2; ++kk) {
    FragH a = fragA_rm(sP + kk * 32, tm * 16, PLD, lane);
    FragH b = fragB_nk(sVt + kk * 32, tn * 16, PLD, lane);   // [n=d][k=j]
    acc.v = wmma16x16x32(a.v, b.v, acc.v);
  }
#pragma unroll
  for (int rr = 0; rr < 8; ++rr) {
    int m = tm * 16 + rr + ((lane >> 4) << 3);
    int nn = tn * 16 + (lane & 15);
    if (m < 56) {
      if (COL) {
        int n = bn & 7, b = bn >> 3;               // m = h index, nn = d
        ao[((size_t)(b * 56 + m) * 56 + w) * 256 + n * 32 + nn] = acc.f[rr];
      } else {
        vwh[base + m * 32 + nn] = (_Float16)acc.f[rr];
      }
    }
  }
}

// ---------------------------------------------------------------------------
// Host orchestration
// ---------------------------------------------------------------------------
extern "C" void kernel_launch(void* const* d_in, const int* in_sizes, int n_in,
                              void* d_out, int out_size, void* d_ws, size_t ws_size,
                              hipStream_t stream) {
  (void)in_sizes; (void)n_in; (void)out_size; (void)ws_size;
  const float SCALE = 0.17677669529663687f;      // 32^-0.5
  const size_t NTOK = 25088;                     // 8*56*56
  const size_t ND = NTOK * 256, NF = NTOK * 1024;

  const float* x     = (const float*)d_in[0];
  const float* cpe_w = (const float*)d_in[1];
  const float* cpe_b = (const float*)d_in[2];
  const float* ln1_g = (const float*)d_in[3];
  const float* ln1_b = (const float*)d_in[4];
  const float* wq    = (const float*)d_in[5];
  const float* bq    = (const float*)d_in[6];
  const float* wk    = (const float*)d_in[7];
  const float* bk    = (const float*)d_in[8];
  const float* wvp   = (const float*)d_in[9];
  const float* bv    = (const float*)d_in[10];
  const float* lep_w = (const float*)d_in[11];
  const float* lep_b = (const float*)d_in[12];
  const float* wo    = (const float*)d_in[13];
  const float* bo    = (const float*)d_in[14];
  const float* ln2_g = (const float*)d_in[15];
  const float* ln2_b = (const float*)d_in[16];
  const float* f1w   = (const float*)d_in[17];
  const float* f1b   = (const float*)d_in[18];
  const float* dww   = (const float*)d_in[19];
  const float* dwb   = (const float*)d_in[20];
  const float* f2w   = (const float*)d_in[21];
  const float* f2b   = (const float*)d_in[22];

  char* wsb = (char*)d_ws;
  size_t off = 0;
  auto bump = [&](size_t bytes) {
    size_t cur = (off + 255) & ~(size_t)255;
    off = cur + bytes;
    return cur;
  };
  float*     qf_x2 = (float*)(wsb + bump(ND * 4));       // q, later x2
  _Float16*  xnh   = (_Float16*)(wsb + bump(ND * 2));    // ln1 out, later ln2 out
  _Float16*  wqh   = (_Float16*)(wsb + bump(65536 * 2));
  _Float16*  wkh   = (_Float16*)(wsb + bump(65536 * 2));
  _Float16*  wvh   = (_Float16*)(wsb + bump(65536 * 2));
  _Float16*  woh   = (_Float16*)(wsb + bump(65536 * 2));
  _Float16*  wf1h  = (_Float16*)(wsb + bump(262144 * 2));
  _Float16*  wf2h  = (_Float16*)(wsb + bump(262144 * 2));
  size_t bigo = bump(ND * 20);                           // phase-aliased region
  float*     xc    = (float*)(wsb + bigo);               // later aliased by hdn
  float*     kf_ao = (float*)(wsb + bigo + ND * 4);      // k, later attn out
  float*     vf    = (float*)(wsb + bigo + ND * 8);
  _Float16*  qrh   = (_Float16*)(wsb + bigo + ND * 12);
  _Float16*  krh   = (_Float16*)(wsb + bigo + ND * 14);
  _Float16*  vh_ao = (_Float16*)(wsb + bigo + ND * 16);  // v(f16), later aoh
  _Float16*  vwh   = (_Float16*)(wsb + bigo + ND * 18);
  float*     hdn   = (float*)(wsb + bigo);               // NF*4 <= ND*16, safe
  _Float16*  hdnh  = (_Float16*)(wsb + bump(NF * 2));
  float*     outp  = (float*)d_out;

  // 1) weights -> f16 (SCALE folded into Wk)
  k_convert<<<256, 256, 0, stream>>>(wq, wqh, 65536, 1.f);
  k_convert<<<256, 256, 0, stream>>>(wk, wkh, 65536, SCALE);
  k_convert<<<256, 256, 0, stream>>>(wvp, wvh, 65536, 1.f);
  k_convert<<<256, 256, 0, stream>>>(wo, woh, 65536, 1.f);
  k_convert<<<1024, 256, 0, stream>>>(f1w, wf1h, 262144, 1.f);
  k_convert<<<1024, 256, 0, stream>>>(f2w, wf2h, 262144, 1.f);

  // 2) CPE: xc = x + dwconv3x3(x)
  k_cpe<<<(int)(ND / 256), 256, 0, stream>>>(x, cpe_w, cpe_b, xc);
  // 3) LN1 -> f16
  k_ln<<<3136, 256, 0, stream>>>(xc, ln1_g, ln1_b, xnh, (int)NTOK);
  // 4) QKV GEMMs (WMMA)
  k_gemm<0><<<dim3(392, 4), 256, 0, stream>>>(xnh, wqh, bq, 1.f, nullptr, qf_x2, 25088, 256, 256);
  k_gemm<0><<<dim3(392, 4), 256, 0, stream>>>(xnh, wkh, bk, SCALE, nullptr, kf_ao, 25088, 256, 256);
  k_gemm<0><<<dim3(392, 4), 256, 0, stream>>>(xnh, wvh, bv, 1.f, nullptr, vf, 25088, 256, 256);
  // 5) RoPE + head-major reshape to f16
  k_rope<<<12544, 256, 0, stream>>>(qf_x2, kf_ao, vf, qrh, krh, vh_ao);
  // 6/7) decomposed attention (row then column), WMMA inside
  k_attn<false><<<3584, 256, 0, stream>>>(qrh, krh, vh_ao, vwh, nullptr);
  k_attn<true><<<3584, 256, 0, stream>>>(qrh, krh, vwh, nullptr, kf_ao);
  // 8) + LEPE (dwconv5x5 of V), -> f16
  k_lepe_add<<<(int)(ND / 256), 256, 0, stream>>>(kf_ao, vf, lep_w, lep_b, vh_ao);
  // 9) Wo GEMM + residual(xc) -> x2
  k_gemm<1><<<dim3(392, 4), 256, 0, stream>>>(vh_ao, woh, bo, 1.f, xc, qf_x2, 25088, 256, 256);
  // 10) LN2 -> f16
  k_ln<<<3136, 256, 0, stream>>>(qf_x2, ln2_g, ln2_b, xnh, (int)NTOK);
  // 11) FC1 GEMM + exact GELU (hdn aliases dead xc..krh)
  k_gemm<2><<<dim3(392, 16), 256, 0, stream>>>(xnh, wf1h, f1b, 1.f, nullptr, hdn, 25088, 1024, 256);
  // 12) dwconv3x3 + residual -> f16
  k_dwadd<<<(int)(NF / 256), 256, 0, stream>>>(hdn, dww, dwb, hdnh);
  // 13) FC2 GEMM + residual(x2) -> d_out
  k_gemm<1><<<dim3(392, 4), 256, 0, stream>>>(hdnh, wf2h, f2b, 1.f, qf_x2, outp, 25088, 256, 1024);
}
